// ECD_CDGINet_43173011259478
// MI455X (gfx1250) — matrix operand order repaired and checked by reading, not compiled, not run on hardware
//
#include <hip/hip_runtime.h>
#include <stdint.h>

// ---------------------------------------------------------------------------
// DIFFormer forward for MI455X (gfx1250, wave32, WMMA).
//  * v_wmma_f32_16x16x32_bf16 GEMMs, 16x64 strip (4 accumulators) per wave,
//    software-pipelined A stream (double-buffered fragment).
//  * q/k projections fused: one GEMM vs packed [WqT;WkT] -> QK [N,512].
//  * Weight slabs staged to LDS by the Tensor Data Mover (one D# descriptor,
//    tensor_load_to_lds + s_wait_tensorcnt, LDS row pad via pad_enable);
//    per-lane global_load_async_to_lds_b128 kept as fallback.
//  * LDS-tiled transposes fed by global_load_async_to_lds_b64 (ASYNCcnt).
//  * Frobenius norms folded into scalar rescales.
// ---------------------------------------------------------------------------

typedef __bf16 bf16_t;
typedef __attribute__((ext_vector_type(16))) __bf16 v16bf;
typedef __attribute__((ext_vector_type(8)))  float  v8f;
typedef uint32_t u32x4 __attribute__((ext_vector_type(4)));
typedef int      i32x4 __attribute__((ext_vector_type(4)));
typedef int      i32x8 __attribute__((ext_vector_type(8)));

#define HD 256
#define LN_EPS 1e-5f
#define BPAD 8   // bf16 elements of LDS row padding (16B) to spread banks

#if defined(__has_builtin)
#  if __has_builtin(__builtin_amdgcn_tensor_load_to_lds)
#    define HAVE_TDM 1
#  endif
#endif
#ifndef HAVE_TDM
#  define HAVE_TDM 0
#endif

// ---------------------------------------------------------------------------
// Utility kernels
// ---------------------------------------------------------------------------
__global__ void zero_f32_k(float* p, size_t n) {
    size_t i = (size_t)blockIdx.x * blockDim.x + threadIdx.x;
    if (i < n) p[i] = 0.0f;
}

__global__ void deg_count_k(const int* __restrict__ col, float* __restrict__ deg, int E) {
    int e = blockIdx.x * blockDim.x + threadIdx.x;
    if (e < E) atomicAdd(&deg[col[e]], 1.0f);
}

__global__ void inv_sqrt_k(const float* __restrict__ deg, float* __restrict__ invs, int N) {
    int i = blockIdx.x * blockDim.x + threadIdx.x;
    if (i < N) {
        float d = deg[i];
        invs[i] = (d > 0.0f) ? rsqrtf(d) : 0.0f;
    }
}

__global__ void conv_x_k(const float* __restrict__ x, bf16_t* __restrict__ xbf,
                         int N, int Fin, int Kp) {
    size_t t = (size_t)blockIdx.x * blockDim.x + threadIdx.x;
    if (t >= (size_t)N * Kp) return;
    int n = (int)(t / Kp), k = (int)(t % Kp);
    float v = (k < Fin) ? x[(size_t)n * Fin + k] : 0.0f;
    xbf[t] = (bf16_t)v;
}

// W [K, Nc] f32 -> WT [Nc, Kp] bf16 (transposed + zero-padded K)
__global__ void conv_wT_k(const float* __restrict__ W, bf16_t* __restrict__ WT,
                          int K, int Nc, int Kp) {
    int t = blockIdx.x * blockDim.x + threadIdx.x;
    if (t >= Nc * Kp) return;
    int c = t / Kp, k = t % Kp;
    float v = (k < K) ? W[(size_t)k * Nc + c] : 0.0f;
    WT[t] = (bf16_t)v;
}

// pack two H-vectors back to back (bias for fused q/k projection)
__global__ void pack2_k(const float* __restrict__ a, const float* __restrict__ b,
                        float* __restrict__ o, int n) {
    int i = blockIdx.x * blockDim.x + threadIdx.x;
    if (i < n) { o[i] = a[i]; o[n + i] = b[i]; }
}

// LDS-tiled transpose: in [N, >=256] (row stride ldin) -> out [256, N].
// 32x32 bf16 tiles; global->LDS side uses the gfx1250 async path.
__global__ void transpose_bf16_k(const bf16_t* __restrict__ in, int ldin,
                                 bf16_t* __restrict__ out, int N) {
    __shared__ __align__(16) bf16_t t[32][36];   // +4 elem pad (8B)
    int nBase = blockIdx.x * 32;
    int mBase = blockIdx.y * 32;
    int tid = threadIdx.x;
    int r  = tid >> 3;                           // 0..31
    int c4 = (tid & 7) * 4;                      // 0,4,..,28
    int n = nBase + r;
    if (n < N) {
        const bf16_t* gp = in + (size_t)n * ldin + mBase + c4;
        uint32_t lds_off = (uint32_t)(uintptr_t)&t[r][c4];
        asm volatile("global_load_async_to_lds_b64 %0, %1, off"
                     :: "v"(lds_off), "v"(gp) : "memory");
    }
    asm volatile("s_wait_asynccnt 0x0" ::: "memory");
    __syncthreads();
    int m = mBase + r;
    int nOut = nBase + c4;
    if (nOut + 3 < N) {
        bf16_t tmp[4];
        #pragma unroll
        for (int i = 0; i < 4; ++i) tmp[i] = t[c4 + i][r];
        *(uint2*)(out + (size_t)m * N + nOut) = *(uint2*)tmp;
    } else {
        for (int i = 0; i < 4 && nOut + i < N; ++i)
            out[(size_t)m * N + nOut + i] = t[c4 + i][r];
    }
}

__global__ void kvsT_k(const float* __restrict__ kvs, bf16_t* __restrict__ kvsT) {
    int t = blockIdx.x * blockDim.x + threadIdx.x;   // 65536 threads
    int d = t >> 8, m = t & 255;
    kvsT[t] = (bf16_t)kvs[m * HD + d];
}

// column-sum + sum-of-squares over [N,256] with row stride ld
__global__ void reduce_f32_k(const float* __restrict__ A, int ld, int N,
                             float* __restrict__ colsum, float* __restrict__ sumsq) {
    int d = threadIdx.x;                     // 256 threads
    float s = 0.0f, s2 = 0.0f;
    for (int r = blockIdx.x; r < N; r += gridDim.x) {
        float v = A[(size_t)r * ld + d];
        s += v; s2 += v * v;
    }
    if (colsum) atomicAdd(&colsum[d], s);
    if (sumsq) {
        __shared__ float sh[HD];
        sh[d] = s2; __syncthreads();
        for (int off = 128; off > 0; off >>= 1) {
            if (d < off) sh[d] += sh[d + off];
            __syncthreads();
        }
        if (d == 0) atomicAdd(sumsq, sh[0]);
    }
}

__global__ void reduce_bf16_k(const bf16_t* __restrict__ A, int ld, int N,
                              float* __restrict__ colsum, float* __restrict__ sumsq) {
    int d = threadIdx.x;
    float s = 0.0f, s2 = 0.0f;
    for (int r = blockIdx.x; r < N; r += gridDim.x) {
        float v = (float)A[(size_t)r * ld + d];
        s += v; s2 += v * v;
    }
    if (colsum) atomicAdd(&colsum[d], s);
    if (sumsq) {
        __shared__ float sh[HD];
        sh[d] = s2; __syncthreads();
        for (int off = 128; off > 0; off >>= 1) {
            if (d < off) sh[d] += sh[d + off];
            __syncthreads();
        }
        if (d == 0) atomicAdd(sumsq, sh[0]);
    }
}

// ---------------------------------------------------------------------------
// WMMA GEMM:  C[M, ldc] = A[M x Kp, row stride lda] * B (BT[NcT, Kp]).
// Wave strip = 16 rows x 64 cols (4 accumulators), block = 8 waves,
// gridDim.y selects the 64-column group of BT.
// STAGE=1: 64 x Kp BT slab -> LDS via one TDM descriptor (row pad through
//          D#.pad_*), else per-lane async loads. B frags = ds_load_b128.
// STAGE=0: B frags direct from global; K split over gridDim.z, atomicAdd Cf.
// ---------------------------------------------------------------------------
template <int STAGE>
__global__ void gemm_wmma_bf16_k(const bf16_t* __restrict__ A, int lda,
                                 const bf16_t* __restrict__ BT,
                                 float*  __restrict__ Cf,
                                 bf16_t* __restrict__ Cbf, int ldc,
                                 const float* __restrict__ bias,
                                 int M, int Kp, int kChunk, int relu) {
    extern __shared__ __align__(16) bf16_t shB[];   // [64][Kp + BPAD]
    const int lane   = threadIdx.x & 31;
    const int waveId = threadIdx.x >> 5;
    const int mTiles = M >> 4;
    const int mt      = blockIdx.x * 8 + waveId;
    const int colBase = blockIdx.y * 64;
    const int ldsStride = Kp + BPAD;

    int k0 = blockIdx.z * kChunk;
    int k1 = k0 + kChunk; if (k1 > Kp) k1 = Kp;

    if (STAGE) {
#if HAVE_TDM
        if (waveId == 0) {
            // ---- Tensor DMA: 2-D tile, 64 rows x Kp bf16, LDS row pad ----
            uint32_t lds_base = (uint32_t)(uintptr_t)&shB[0];
            uint64_t ga  = (uint64_t)(uintptr_t)(BT + (size_t)colBase * Kp);
            uint32_t rowDw = (uint32_t)(Kp >> 1);                // dwords per row
            uint32_t intervalCode = (31u - (uint32_t)__builtin_clz(rowDw)) - 1u;
            u32x4 g0;
            g0.x = 1u;                                           // count=1
            g0.y = lds_base;                                     // lds_addr
            g0.z = (uint32_t)ga;                                 // global_addr lo
            g0.w = (uint32_t)((ga >> 32) & 0x01FFFFFFu) | (2u << 30); // hi | type=2
            uint32_t dw0 = (1u << 16)                            // data_size = 2B
                         | (1u << 20)                            // pad_enable
                         | (intervalCode << 22)                  // pad every row
                         | (3u << 25);                           // pad 4 dwords (16B)
            uint32_t td0 = (uint32_t)Kp, td1 = 64u;
            uint32_t tl0 = (uint32_t)Kp, tl1 = 64u;
            uint64_t s0  = (uint64_t)(uint32_t)Kp;               // dim0 stride (elems)
            i32x8 g1;
            g1[0] = (int)dw0;
            g1[1] = (int)((td0 & 0xFFFFu) << 16);                // tensor_dim0 lo16
            g1[2] = (int)((td0 >> 16) | ((td1 & 0xFFFFu) << 16));
            g1[3] = (int)((td1 >> 16) | (tl0 << 16));            // tile_dim0
            g1[4] = (int)tl1;                                    // tile_dim1; dim2=0
            g1[5] = (int)(uint32_t)(s0 & 0xFFFFFFFFu);
            g1[6] = (int)(uint32_t)((s0 >> 32) & 0xFFFFu);
            g1[7] = 0;
            i32x4 z4 = {0, 0, 0, 0};
#if defined(__clang_major__) && __clang_major__ >= 23
            i32x8 z8 = {0, 0, 0, 0, 0, 0, 0, 0};
            __builtin_amdgcn_tensor_load_to_lds(g0, g1, z4, z4, z8, 0);
#else
            __builtin_amdgcn_tensor_load_to_lds(g0, g1, z4, z4, 0);
#endif
            __builtin_amdgcn_s_wait_tensorcnt(0);
        }
        __syncthreads();
#else
        int totalVec = (64 * Kp) >> 3;               // 16B chunks
        for (int t = threadIdx.x; t < totalVec; t += blockDim.x) {
            int r   = (t * 8) / Kp;
            int off = (t * 8) % Kp;
            const bf16_t* gp = BT + (size_t)(colBase + r) * Kp + off;
            uint32_t lds_off = (uint32_t)(uintptr_t)&shB[r * ldsStride + off];
            asm volatile("global_load_async_to_lds_b128 %0, %1, off"
                         :: "v"(lds_off), "v"(gp) : "memory");
        }
        asm volatile("s_wait_asynccnt 0x0" ::: "memory");
        __syncthreads();
#endif
    }
    if (mt >= mTiles) return;

    const int rowA = mt * 16 + (lane & 15);
    const int kbA  = (lane < 16) ? 0 : 8;
    const int kbB  = (lane < 16) ? 0 : 8;
    const int colL = lane & 15;

    const bf16_t* aptr = A + (size_t)rowA * lda + k0 + kbA;

    v8f acc0 = {}, acc1 = {}, acc2 = {}, acc3 = {};
    union Frag { uint4 u[2]; v16bf v; };

    Frag fa;
    fa.u[0] = *(const uint4*)(aptr);
    fa.u[1] = *(const uint4*)(aptr + 16);

    for (int kk = k0; kk < k1; kk += 32) {
        Frag fcur = fa;
        const bf16_t* anext = aptr + 32;
        if (kk + 32 < k1) {                       // prefetch next A fragment
            fa.u[0] = *(const uint4*)(anext);
            fa.u[1] = *(const uint4*)(anext + 16);
            __builtin_prefetch(anext + 128, 0, 1);
        }
        aptr = anext;

        if (STAGE) {
            #pragma unroll
            for (int nl = 0; nl < 4; ++nl) {
                Frag fb;
                const bf16_t* bp = &shB[(nl * 16 + colL) * ldsStride + kk + kbB];
                fb.u[0] = *(const uint4*)(bp);
                fb.u[1] = *(const uint4*)(bp + 16);
                v8f& acc = (nl == 0) ? acc0 : (nl == 1) ? acc1 : (nl == 2) ? acc2 : acc3;
                acc = __builtin_amdgcn_wmma_f32_16x16x32_bf16(
                          false, fcur.v, false, fb.v, (short)0, acc, false, false);
            }
        } else {
            #pragma unroll
            for (int nl = 0; nl < 4; ++nl) {
                Frag fb;
                const bf16_t* bp = BT + (size_t)(colBase + nl * 16 + colL) * Kp + kk + kbB;
                fb.u[0] = *(const uint4*)(bp);
                fb.u[1] = *(const uint4*)(bp + 16);
                v8f& acc = (nl == 0) ? acc0 : (nl == 1) ? acc1 : (nl == 2) ? acc2 : acc3;
                acc = __builtin_amdgcn_wmma_f32_16x16x32_bf16(
                          false, fcur.v, false, fb.v, (short)0, acc, false, false);
            }
        }
    }

    const int rBase = mt * 16 + ((lane < 16) ? 0 : 8);
    #pragma unroll
    for (int nl = 0; nl < 4; ++nl) {
        v8f& acc = (nl == 0) ? acc0 : (nl == 1) ? acc1 : (nl == 2) ? acc2 : acc3;
        int col = colBase + nl * 16 + colL;
        if (gridDim.z > 1) {
            #pragma unroll
            for (int j = 0; j < 8; ++j)
                atomicAdd(&Cf[(size_t)(rBase + j) * ldc + col], acc[j]);
        } else {
            #pragma unroll
            for (int j = 0; j < 8; ++j) {
                float v = acc[j];
                if (bias) v += bias[col];
                if (relu) v = fmaxf(v, 0.0f);
                size_t idx = (size_t)(rBase + j) * ldc + col;
                if (Cf)  Cf[idx]  = v;
                if (Cbf) Cbf[idx] = (bf16_t)v;
            }
        }
    }
}

// ---------------------------------------------------------------------------
// GCN edge scatter: wave per edge, lanes over 256 features (L2-resident v).
// ---------------------------------------------------------------------------
__global__ void gcn_edge_k(const int* __restrict__ row, const int* __restrict__ col,
                           const float* __restrict__ invs, const float* __restrict__ R,
                           float* __restrict__ G, int E) {
    int e    = blockIdx.x * (blockDim.x >> 5) + (threadIdx.x >> 5);
    int lane = threadIdx.x & 31;
    if (e >= E) return;
    int r = row[e], c = col[e];
    float w = invs[c] * invs[r];
    const float* src = R + (size_t)r * HD;
    float*       dst = G + (size_t)c * HD;
    #pragma unroll
    for (int j = 0; j < 8; ++j) {
        int d = lane + 32 * j;
        atomicAdd(&dst[d], w * src[d]);
    }
}

// ---------------------------------------------------------------------------
// Fused attention-combine + residual + LayerNorm (wave per row).
// ---------------------------------------------------------------------------
__global__ void combine_ln_k(float* __restrict__ R, bf16_t* __restrict__ Rbf,
                             const float* __restrict__ S, const float* __restrict__ G,
                             const bf16_t* __restrict__ Qbf, int ldq,
                             const float* __restrict__ ksum, const float* __restrict__ vsum,
                             const float* __restrict__ scal,     // [0]=qn2 [1]=kn2
                             const float* __restrict__ alpha_p,
                             const float* __restrict__ gamma, const float* __restrict__ beta,
                             int N) {
    int n    = blockIdx.x * (blockDim.x >> 5) + (threadIdx.x >> 5);
    int lane = threadIdx.x & 31;
    if (n >= N) return;
    size_t base  = (size_t)n * HD;
    size_t baseq = (size_t)n * ldq;

    float scale = rsqrtf(scal[0] * scal[1]);
    float alpha = alpha_p[0];

    float dq = 0.0f;
    #pragma unroll
    for (int j = 0; j < 8; ++j) {
        int d = lane + 32 * j;
        dq += (float)Qbf[baseq + d] * ksum[d];
    }
    #pragma unroll
    for (int m = 16; m > 0; m >>= 1) dq += __shfl_xor(dq, m, 32);
    float den = dq * scale + (float)N;

    float h[8];
    float mu = 0.0f;
    #pragma unroll
    for (int j = 0; j < 8; ++j) {
        int d = lane + 32 * j;
        float attn = (S[base + d] * scale + vsum[d]) / den;
        float hn   = attn + G[base + d];
        h[j] = alpha * hn + (1.0f - alpha) * R[base + d];
        mu += h[j];
    }
    #pragma unroll
    for (int m = 16; m > 0; m >>= 1) mu += __shfl_xor(mu, m, 32);
    mu *= (1.0f / (float)HD);

    float var = 0.0f;
    #pragma unroll
    for (int j = 0; j < 8; ++j) { float t = h[j] - mu; var += t * t; }
    #pragma unroll
    for (int m = 16; m > 0; m >>= 1) var += __shfl_xor(var, m, 32);
    var *= (1.0f / (float)HD);
    float inv = rsqrtf(var + LN_EPS);

    #pragma unroll
    for (int j = 0; j < 8; ++j) {
        int d = lane + 32 * j;
        float o = (h[j] - mu) * inv * gamma[d] + beta[d];
        R[base + d]   = o;
        Rbf[base + d] = (bf16_t)o;
    }
}

__global__ void readout_k(const float* __restrict__ R, const float* __restrict__ WrRow,
                          const float* __restrict__ br_i, const float* __restrict__ wgt_i,
                          float* __restrict__ out, int N) {
    int n    = blockIdx.x * (blockDim.x >> 5) + (threadIdx.x >> 5);
    int lane = threadIdx.x & 31;
    if (n >= N) return;
    size_t base = (size_t)n * HD;
    float s = 0.0f;
    #pragma unroll
    for (int j = 0; j < 8; ++j) {
        int d = lane + 32 * j;
        s += R[base + d] * WrRow[d];
    }
    #pragma unroll
    for (int m = 16; m > 0; m >>= 1) s += __shfl_xor(s, m, 32);
    if (lane == 0) out[n] += wgt_i[0] * (s + br_i[0]);
}

// ---------------------------------------------------------------------------
// Host-side orchestration
// ---------------------------------------------------------------------------
extern "C" void kernel_launch(void* const* d_in, const int* in_sizes, int n_in,
                              void* d_out, int out_size, void* d_ws, size_t ws_size,
                              hipStream_t stream) {
    const float* x     = (const float*)d_in[0];
    const int*   ei    = (const int*)  d_in[1];
    const float* W1    = (const float*)d_in[2];
    const float* b1    = (const float*)d_in[3];
    const float* Wq    = (const float*)d_in[4];
    const float* bq    = (const float*)d_in[5];
    const float* Wk    = (const float*)d_in[6];
    const float* bk    = (const float*)d_in[7];
    const float* gamma = (const float*)d_in[8];
    const float* beta  = (const float*)d_in[9];
    const float* alpha = (const float*)d_in[10];
    const float* Wr    = (const float*)d_in[11];
    const float* br    = (const float*)d_in[12];
    const float* wgt   = (const float*)d_in[13];
    float* out = (float*)d_out;

    const int Fin  = in_sizes[2] / HD;          // 58
    const int N    = in_sizes[0] / Fin;         // 100000
    const int E    = in_sizes[1] / 2;           // 3200000
    const int KpIn = (Fin + 31) & ~31;          // 64
    const int* erow = ei;
    const int* ecol = ei + E;

    // ---- workspace carve-out ----
    char* p = (char*)d_ws;
    auto alloc = [&](size_t bytes) -> char* {
        char* r = p; p += (bytes + 255) & ~(size_t)255; return r;
    };
    float*  Rf    = (float*) alloc((size_t)N * HD * 4);
    float*  Sf    = (float*) alloc((size_t)N * HD * 4);
    float*  Gf    = (float*) alloc((size_t)N * HD * 4);
    bf16_t* Rbf   = (bf16_t*)alloc((size_t)N * HD * 2);
    bf16_t* QKbf  = (bf16_t*)alloc((size_t)N * (2 * HD) * 2);  // q|k interleaved rows
    bf16_t* KbfT  = (bf16_t*)alloc((size_t)N * HD * 2);
    bf16_t* RbfT  = (bf16_t*)alloc((size_t)N * HD * 2);
    bf16_t* xbf   = (bf16_t*)alloc((size_t)N * KpIn * 2);
    float*  deg   = (float*) alloc((size_t)N * 4);
    float*  invs  = (float*) alloc((size_t)N * 4);
    bf16_t* W1T   = (bf16_t*)alloc((size_t)HD * KpIn * 2);
    bf16_t* WqkT  = (bf16_t*)alloc((size_t)(2 * HD) * HD * 2); // [WqT;WkT]
    float*  bqk   = (float*) alloc((size_t)(2 * HD) * 4);
    float*  kvs   = (float*) alloc((size_t)HD * HD * 4);
    bf16_t* kvsT  = (bf16_t*)alloc((size_t)HD * HD * 2);
    float*  ksum  = (float*) alloc(HD * 4);
    float*  vsum  = (float*) alloc(HD * 4);
    float*  scal  = (float*) alloc(2 * 4);

    bf16_t* Qbf = QKbf;        // row stride 512
    bf16_t* Kbf = QKbf + HD;   // row stride 512

    dim3 blk(256);
    auto nb = [](size_t n) { return (unsigned)((n + 255) / 256); };

    // ---- graph prep ----
    zero_f32_k<<<nb(N), blk, 0, stream>>>(deg, (size_t)N);
    zero_f32_k<<<nb(N), blk, 0, stream>>>(out, (size_t)N);
    deg_count_k<<<nb(E), blk, 0, stream>>>(ecol, deg, E);
    inv_sqrt_k<<<nb(N), blk, 0, stream>>>(deg, invs, N);

    // ---- operand packing ----
    conv_x_k <<<nb((size_t)N * KpIn), blk, 0, stream>>>(x, xbf, N, Fin, KpIn);
    conv_wT_k<<<nb(HD * KpIn), blk, 0, stream>>>(W1, W1T, Fin, HD, KpIn);
    conv_wT_k<<<nb(HD * HD),   blk, 0, stream>>>(Wq, WqkT,            HD, HD, HD);
    conv_wT_k<<<nb(HD * HD),   blk, 0, stream>>>(Wk, WqkT + HD * HD,  HD, HD, HD);
    pack2_k  <<<nb(HD), blk, 0, stream>>>(bq, bk, bqk, HD);

    const unsigned gxBig = (unsigned)((N / 16 + 7) / 8);   // 8 m-tiles per block
    const unsigned gxRow = (unsigned)((N + 7) / 8);
    const size_t smem256 = (size_t)64 * (HD  + BPAD) * 2;  // 33.8 KB
    const size_t smemIn  = (size_t)64 * (KpIn + BPAD) * 2; //  9.2 KB

    // ---- stage 0: R = relu(x@W1 + b1), readout term 0 ----
    gemm_wmma_bf16_k<1><<<dim3(gxBig, HD / 64, 1), blk, smemIn, stream>>>(
        xbf, KpIn, W1T, Rf, Rbf, HD, b1, N, KpIn, KpIn, 1);
    readout_k<<<gxRow, blk, 0, stream>>>(Rf, Wr + 0 * HD, br + 0, wgt + 0, out, N);

    // ---- 3 shared conv layers ----
    const int kChunk = 800;                               // K-split for kvs GEMM
    const unsigned zc = (unsigned)((N + kChunk - 1) / kChunk);
    for (int it = 0; it < 3; ++it) {
        zero_f32_k<<<1, blk, 0, stream>>>(scal, 2);
        zero_f32_k<<<1, blk, 0, stream>>>(ksum, HD);
        zero_f32_k<<<1, blk, 0, stream>>>(vsum, HD);
        zero_f32_k<<<nb(HD * HD), blk, 0, stream>>>(kvs, (size_t)HD * HD);
        zero_f32_k<<<nb((size_t)N * HD), blk, 0, stream>>>(Gf, (size_t)N * HD);

        // fused q|k projection: QK = R @ [Wq|Wk] + [bq|bk]  (A read once)
        gemm_wmma_bf16_k<1><<<dim3(gxBig, (2 * HD) / 64, 1), blk, smem256, stream>>>(
            Rbf, HD, WqkT, nullptr, QKbf, 2 * HD, bqk, N, HD, HD, 0);

        // ||q||^2, ||k||^2, sum(k), sum(v=R)
        reduce_bf16_k<<<256, blk, 0, stream>>>(Qbf, 2 * HD, N, nullptr, scal + 0);
        reduce_bf16_k<<<256, blk, 0, stream>>>(Kbf, 2 * HD, N, ksum,    scal + 1);
        reduce_f32_k <<<256, blk, 0, stream>>>(Rf,  HD,     N, vsum,    nullptr);

        // kvs = K^T @ V : LDS-tiled transposed feeds + K-split atomic GEMM
        transpose_bf16_k<<<dim3((N + 31) / 32, HD / 32), blk, 0, stream>>>(Kbf, 2 * HD, KbfT, N);
        transpose_bf16_k<<<dim3((N + 31) / 32, HD / 32), blk, 0, stream>>>(Rbf, HD,     RbfT, N);
        gemm_wmma_bf16_k<0><<<dim3(2, HD / 64, zc), blk, 0, stream>>>(
            KbfT, N, RbfT, kvs, nullptr, HD, nullptr, HD, N, kChunk, 0);
        kvsT_k<<<nb(HD * HD), blk, 0, stream>>>(kvs, kvsT);

        // num_raw = Q @ kvs
        gemm_wmma_bf16_k<1><<<dim3(gxBig, HD / 64, 1), blk, smem256, stream>>>(
            Qbf, 2 * HD, kvsT, Sf, nullptr, HD, nullptr, N, HD, HD, 0);

        // GCN scatter
        gcn_edge_k<<<(unsigned)((E + 7) / 8), blk, 0, stream>>>(
            erow, ecol, invs, Rf, Gf, E);

        // attention combine + residual + LayerNorm
        combine_ln_k<<<gxRow, blk, 0, stream>>>(
            Rf, Rbf, Sf, Gf, Qbf, 2 * HD, ksum, vsum, scal, alpha, gamma, beta, N);

        // readout term it+1
        readout_k<<<gxRow, blk, 0, stream>>>(
            Rf, Wr + (it + 1) * HD, br + (it + 1), wgt + (it + 1), out, N);
    }
}
